// ManualGRU_74466142978186
// MI455X (gfx1250) — matrix operand order repaired and checked by reading, not compiled, not run on hardware
//
#include <hip/hip_runtime.h>
#include <hip/hip_bf16.h>
#include <math.h>

// ---------------------------------------------------------------------------
// GRU (persistent WMMA kernel, bf16 matmul / f32 accumulate) + fused
// projection + log_softmax for gfx1250 (CDNA5, wave32, WMMA).
//
// Data movement: all WMMA operand staging is 16B-chunk copies from bf16
// global mirrors, via GLOBAL_LOAD_ASYNC_TO_LDS_B128 when the builtin exists
// (ASYNCcnt path), else plain b128 copies.
//
// Workspace layout (bytes, 256-aligned cursor):
//   0          ctr      u32[2] arrive/release     (256)
//   256        h        f32 [64][512]             (131072)
//   131328     z        f32 [64][512]             (131072)
//   262400     hbf      u16 [64][512]             (65536)
//   327936     rhbf     u16 [64][512]             (65536)
//   393472     xbf      u16 [1024][64][256]       (33554432)
//   33947904   Wo_bf    u16 [1024][512]           (1048576)
//   34996480   hs_bf    u16 [1024][64][512]       (67108864)
//   total ~102.1 MB
// ---------------------------------------------------------------------------

#define T_STEPS 1024
#define BATCH   64
#define IDIM    256
#define HDIM    512
#define ODIM    1024
#define KDIM    768      // H + I
#define KPAD    776      // +8 bf16 pad -> row stride 1552B, bank-friendly
#define NWG     16       // persistent workgroups; each owns 32 columns of H
#define HPAD    520      // hs tile pad in projection kernel

typedef __attribute__((ext_vector_type(16))) __bf16 v16bf;
typedef __attribute__((ext_vector_type(8)))  float  v8f;
typedef __attribute__((ext_vector_type(4)))  unsigned int u32x4;
typedef __attribute__((ext_vector_type(4)))  int          i32x4;
typedef unsigned short u16;

struct Frag32B { u32x4 lo, hi; };

#if __has_builtin(__builtin_amdgcn_global_load_async_to_lds_b128)
#define HAVE_ASYNC_LDS 1
typedef i32x4 __attribute__((address_space(1))) *gv4p;   // global int4*
typedef i32x4 __attribute__((address_space(3))) *lv4p;   // lds int4*
#else
#define HAVE_ASYNC_LDS 0
#endif

// 16-byte global(bf16) -> LDS copy.
__device__ inline void copy16(u16* ldsDst, const u16* glbSrc) {
#if HAVE_ASYNC_LDS
    __builtin_amdgcn_global_load_async_to_lds_b128(
        (gv4p)glbSrc, (lv4p)ldsDst, 0, 0);
#else
    *(u32x4*)ldsDst = *(const u32x4*)glbSrc;
#endif
}

__device__ inline void async_join() {
#if HAVE_ASYNC_LDS
#if __has_builtin(__builtin_amdgcn_s_wait_asynccnt)
    __builtin_amdgcn_s_wait_asynccnt(0);
#else
    asm volatile("s_wait_asynccnt 0" ::: "memory");
#endif
#endif
}

__device__ inline u16 f2bf(float f) {
    unsigned u = __builtin_bit_cast(unsigned, f);
    u += 0x7FFFu + ((u >> 16) & 1u);      // round to nearest even
    return (u16)(u >> 16);
}

// A fragment: 16(M) x 32(K) bf16. lane: m = lane&15, khalf = lane>>4.
__device__ inline v16bf load_a(const u16* base, int row, int rowStride,
                               int kbase, int khalf) {
    const u16* p = base + row * rowStride + kbase + khalf * 16;
    Frag32B t;
    t.lo = *(const u32x4*)(p);
    t.hi = *(const u32x4*)(p + 8);
    return __builtin_bit_cast(v16bf, t);
}

// B fragment: 32(K) x 16(N) bf16 from row-major W[n][k].
__device__ inline v16bf load_b_lds(const u16* base, int ncol, int rowStride,
                                   int kbase, int khalf) {
    const u16* p = base + ncol * rowStride + kbase + khalf * 16;
    Frag32B t;
    t.lo = *(const u32x4*)(p);
    t.hi = *(const u32x4*)(p + 8);
    return __builtin_bit_cast(v16bf, t);
}

__device__ inline v16bf load_b_glb(const u16* base, int ncol, int rowStride,
                                   int kbase, int khalf) {
    const u16* p = base + (size_t)ncol * rowStride + kbase + khalf * 16;
    Frag32B t;
    t.lo = *(const u32x4*)(p);
    t.hi = *(const u32x4*)(p + 8);
    return __builtin_bit_cast(v16bf, t);
}

__device__ inline v8f wmma_bf16(v16bf a, v16bf b, v8f c) {
    return __builtin_amdgcn_wmma_f32_16x16x32_bf16(
        false, a, false, b, (short)0, c, false, false);
}

// Device-wide barrier: monotonic arrive counter + release generation.
__device__ inline void grid_sync(unsigned* arrive, unsigned* release,
                                 unsigned gen) {
    __syncthreads();
    if (threadIdx.x == 0) {
        unsigned prev = __hip_atomic_fetch_add(arrive, 1u, __ATOMIC_ACQ_REL,
                                               __HIP_MEMORY_SCOPE_AGENT);
        if (prev + 1u == gen * NWG) {
            __hip_atomic_store(release, gen, __ATOMIC_RELEASE,
                               __HIP_MEMORY_SCOPE_AGENT);
        }
        while (__hip_atomic_load(release, __ATOMIC_ACQUIRE,
                                 __HIP_MEMORY_SCOPE_AGENT) < gen) {
            __builtin_amdgcn_s_sleep(8);
        }
    }
    __syncthreads();
}

// ---------------------------------------------------------------------------
// Persistent GRU kernel: 16 WGs x 256 threads (8 wave32). WG g owns hidden
// columns [g*32, g*32+32). Gate weights for that slice live in LDS (bf16) for
// all 1024 steps. Two device barriers per step.
// ---------------------------------------------------------------------------
__global__ void gru_persistent(const u16* __restrict__ xbf,
                               float* __restrict__ h,
                               float* __restrict__ zb,
                               u16* __restrict__ hbf,
                               u16* __restrict__ rhbf,
                               const float* __restrict__ Wr,
                               const float* __restrict__ Wz,
                               const float* __restrict__ Wn,
                               const float* __restrict__ br,
                               const float* __restrict__ bz,
                               const float* __restrict__ bn,
                               u16* __restrict__ hs,
                               unsigned* arrive, unsigned* release) {
    extern __shared__ char smem[];
    u16* wsl  = (u16*)smem;                                 // [3][32][KPAD]
    u16* comb = (u16*)(smem + 3 * 32 * KPAD * sizeof(u16)); // [64][KPAD]

    const int tid   = threadIdx.x;
    const int wave  = tid >> 5;
    const int lane  = tid & 31;
    const int khalf = lane >> 4;
    const int l16   = lane & 15;
    const int g     = blockIdx.x;           // column group
    const int rw0   = wave * 8;             // rows this wave stages

    // --- Load this WG's weight slices (f32 -> bf16) into LDS, once. ---
    for (int gt = 0; gt < 3; ++gt) {
        const float* W = (gt == 0) ? Wr : ((gt == 1) ? Wz : Wn);
        for (int n = 0; n < 32; ++n) {
            const float* src = W + (size_t)(g * 32 + n) * KDIM;
            u16* dst = wsl + (gt * 32 + n) * KPAD;
            for (int c = tid; c < KDIM; c += 256) dst[c] = f2bf(src[c]);
        }
    }
    __syncthreads();

    unsigned gen = 0;
    for (int t = 0; t < T_STEPS; ++t) {
        // --- stage comb = [hbf | xbf_t] via 16B chunk copies (async) ---
        const u16* xt = xbf + (size_t)t * BATCH * IDIM;
        for (int rr = 0; rr < 8; ++rr) {
            const int row = rw0 + rr;
            u16* crow = comb + row * KPAD;
            for (int c = lane; c < 96; c += 32) {   // 96 x 16B per row
                const u16* src = (c < 64) ? (hbf + row * HDIM + c * 8)
                                          : (xt + row * IDIM + (c - 64) * 8);
                copy16(crow + c * 8, src);
            }
        }
        async_join();
        __syncthreads();

        // --- phase 1: r (waves 0-3) and z (waves 4-7), 2 N-tiles each ---
        {
            const int gate  = wave >> 2;         // 0 = r, 1 = z
            const int mtile = wave & 3;
            const int arow  = mtile * 16 + l16;
            v8f acc0 = {}, acc1 = {};
            for (int kk = 0; kk < KDIM / 32; ++kk) {
                const int kb = kk * 32;
                v16bf a  = load_a(comb, arow, KPAD, kb, khalf);
                v16bf b0 = load_b_lds(wsl, gate * 32 + l16,      KPAD, kb, khalf);
                v16bf b1 = load_b_lds(wsl, gate * 32 + 16 + l16, KPAD, kb, khalf);
                acc0 = wmma_bf16(a, b0, acc0);
                acc1 = wmma_bf16(a, b1, acc1);
            }
            const float* bias = (gate == 0) ? br : bz;
            for (int pass = 0; pass < 2; ++pass) {
                v8f acc = pass ? acc1 : acc0;
                const int col = g * 32 + pass * 16 + l16;
                const float bv = bias[col];
                #pragma unroll
                for (int i = 0; i < 8; ++i) {
                    const int rrow = mtile * 16 + i + 8 * khalf;
                    float s = 1.0f / (1.0f + __expf(-(acc[i] + bv)));
                    if (gate == 0)
                        rhbf[rrow * HDIM + col] = f2bf(s * h[rrow * HDIM + col]);
                    else
                        zb[rrow * HDIM + col] = s;
                }
            }
        }
        grid_sync(arrive, release, ++gen);

        // --- refresh comb h-part with r*h (x-part unchanged) ---
        for (int rr = 0; rr < 8; ++rr) {
            const int row = rw0 + rr;
            u16* crow = comb + row * KPAD;
            for (int c = lane; c < 64; c += 32)     // 64 x 16B per row
                copy16(crow + c * 8, rhbf + row * HDIM + c * 8);
        }
        async_join();
        __syncthreads();

        // --- phase 2: n gate, 1 tile per wave; blend to h_new ---
        {
            const int mtile = wave >> 1;
            const int ntl   = wave & 1;
            const int arow  = mtile * 16 + l16;
            v8f acc = {};
            for (int kk = 0; kk < KDIM / 32; ++kk) {
                const int kb = kk * 32;
                v16bf a = load_a(comb, arow, KPAD, kb, khalf);
                v16bf b = load_b_lds(wsl, 2 * 32 + ntl * 16 + l16, KPAD, kb, khalf);
                acc = wmma_bf16(a, b, acc);
            }
            const int col = g * 32 + ntl * 16 + l16;
            const float bv = bn[col];
            u16* hsrow = hs + (size_t)t * BATCH * HDIM;
            #pragma unroll
            for (int i = 0; i < 8; ++i) {
                const int rrow = mtile * 16 + i + 8 * khalf;
                float nv = tanhf(acc[i] + bv);
                float zv = zb[rrow * HDIM + col];
                float hv = h[rrow * HDIM + col];
                float hn = (1.0f - zv) * nv + zv * hv;
                u16 hb = f2bf(hn);
                h[rrow * HDIM + col]      = hn;
                hbf[rrow * HDIM + col]    = hb;
                hsrow[rrow * HDIM + col]  = hb;
            }
        }
        grid_sync(arrive, release, ++gen);
    }
}

// ---------------------------------------------------------------------------
// Projection + fused log_softmax: each WG handles 16 rows x 1024 cols.
// hs tile (16x512 bf16) staged in LDS via async b128; Wo read as bf16 from L2.
// ---------------------------------------------------------------------------
__global__ void proj_logsoftmax(const u16* __restrict__ hs,
                                const u16* __restrict__ woBf,
                                const float* __restrict__ bo,
                                float* __restrict__ out) {
    extern __shared__ char smem[];
    u16*   hsT    = (u16*)smem;                                   // [16][HPAD]
    float* logits = (float*)(smem + 16 * HPAD * sizeof(u16));     // [16][1024]
    float* red    = logits + 16 * ODIM;                           // [256]
    float* rowlse = red + 256;                                    // [16]
    float* rowmax = rowlse + 16;                                  // [16]

    const int tid   = threadIdx.x;
    const int wave  = tid >> 5;
    const int lane  = tid & 31;
    const int khalf = lane >> 4;
    const int l16   = lane & 15;
    const int rowBase = blockIdx.x * 16;

    // stage 16 hs rows (bf16) -> LDS: 16 rows x 64 chunks of 16B
    for (int e = tid; e < 16 * 64; e += 256) {
        const int r = e >> 6, c = e & 63;
        copy16(hsT + r * HPAD + c * 8,
               hs + (size_t)(rowBase + r) * HDIM + c * 8);
    }
    async_join();
    __syncthreads();

    // GEMM: wave w covers N-tiles {w, w+8, ..., w+56}
    v8f acc[8] = {};
    for (int kk = 0; kk < HDIM / 32; ++kk) {
        const int kb = kk * 32;
        v16bf a = load_a(hsT, l16, HPAD, kb, khalf);
        #pragma unroll
        for (int j = 0; j < 8; ++j) {
            const int ncol = (wave + j * 8) * 16 + l16;
            v16bf b = load_b_glb(woBf, ncol, HDIM, kb, khalf);
            acc[j] = wmma_bf16(a, b, acc[j]);
        }
    }
    #pragma unroll
    for (int j = 0; j < 8; ++j) {
        const int col = (wave + j * 8) * 16 + l16;
        const float bv = bo[col];
        #pragma unroll
        for (int i = 0; i < 8; ++i)
            logits[(i + 8 * khalf) * ODIM + col] = acc[j][i] + bv;
    }
    __syncthreads();

    // row-wise max then logsumexp (16 threads per row)
    const int r  = tid >> 4;
    const int sl = tid & 15;
    float mx = -INFINITY;
    for (int c = sl; c < ODIM; c += 16) mx = fmaxf(mx, logits[r * ODIM + c]);
    red[tid] = mx;
    __syncthreads();
    if (sl == 0) {
        float m = red[tid];
        #pragma unroll
        for (int q = 1; q < 16; ++q) m = fmaxf(m, red[tid + q]);
        rowmax[r] = m;
    }
    __syncthreads();
    const float rm = rowmax[r];
    float s = 0.0f;
    for (int c = sl; c < ODIM; c += 16) s += __expf(logits[r * ODIM + c] - rm);
    red[tid] = s;
    __syncthreads();
    if (sl == 0) {
        float ss = 0.0f;
        #pragma unroll
        for (int q = 0; q < 16; ++q) ss += red[tid + q];
        rowlse[r] = rm + __logf(ss);
    }
    __syncthreads();
    for (int e = tid; e < 16 * ODIM; e += 256) {
        const int rr = e >> 10, c = e & (ODIM - 1);
        out[(size_t)(rowBase + rr) * ODIM + c] = logits[rr * ODIM + c] - rowlse[rr];
    }
}

// ---------------------------------------------------------------------------
// Small helpers
// ---------------------------------------------------------------------------
__global__ void copy_f32_kernel(const float* __restrict__ src,
                                float* __restrict__ dst, int n) {
    int i = blockIdx.x * 256 + threadIdx.x;
    if (i < n) dst[i] = src[i];
}

__global__ void cvt_bf16_kernel(const float* __restrict__ src,
                                u16* __restrict__ dst, int n) {
    int i = blockIdx.x * 256 + threadIdx.x;
    if (i < n) dst[i] = f2bf(src[i]);
}

extern "C" void kernel_launch(void* const* d_in, const int* in_sizes, int n_in,
                              void* d_out, int out_size, void* d_ws, size_t ws_size,
                              hipStream_t stream) {
    const float* x   = (const float*)d_in[0];
    const float* h0  = (const float*)d_in[1];
    const float* Wr  = (const float*)d_in[2];
    const float* br  = (const float*)d_in[3];
    const float* Wz  = (const float*)d_in[4];
    const float* bz  = (const float*)d_in[5];
    const float* Wn  = (const float*)d_in[6];
    const float* bn  = (const float*)d_in[7];
    const float* Wo  = (const float*)d_in[8];
    const float* bo  = (const float*)d_in[9];
    float* out = (float*)d_out;

    char* ws = (char*)d_ws;
    unsigned* ctr = (unsigned*)ws;                       // [arrive, release]
    float* h    = (float*)(ws + 256);
    float* zb   = (float*)(ws + 131328);
    u16*   hbf  = (u16*)(ws + 262400);
    u16*   rhbf = (u16*)(ws + 327936);
    u16*   xbf  = (u16*)(ws + 393472);
    u16*   wo   = (u16*)(ws + 33947904);
    u16*   hs   = (u16*)(ws + 34996480);

    (void)hipMemsetAsync(ctr, 0, 256, stream);
    copy_f32_kernel<<<(BATCH * HDIM + 255) / 256, 256, 0, stream>>>(h0, h, BATCH * HDIM);
    cvt_bf16_kernel<<<(BATCH * HDIM + 255) / 256, 256, 0, stream>>>(h0, hbf, BATCH * HDIM);
    cvt_bf16_kernel<<<(T_STEPS * BATCH * IDIM + 255) / 256, 256, 0, stream>>>(
        x, xbf, T_STEPS * BATCH * IDIM);
    cvt_bf16_kernel<<<(ODIM * HDIM + 255) / 256, 256, 0, stream>>>(Wo, wo, ODIM * HDIM);

    const size_t smem1 = (size_t)3 * 32 * KPAD * sizeof(u16)   // weights  ~145.5KB
                       + (size_t)BATCH * KPAD * sizeof(u16);   // comb     ~97KB
    gru_persistent<<<NWG, 256, smem1, stream>>>(xbf, h, zb, hbf, rhbf,
                                                Wr, Wz, Wn, br, bz, bn,
                                                hs, ctr, ctr + 1);

    const size_t smem2 = (size_t)16 * HPAD * sizeof(u16)       // hs tile
                       + (size_t)16 * ODIM * sizeof(float)     // logits
                       + (256 + 16 + 16) * sizeof(float);      // reductions
    proj_logsoftmax<<<(T_STEPS * BATCH) / 16, 256, smem2, stream>>>(hs, wo, bo, out);

    copy_f32_kernel<<<(BATCH * HDIM + 255) / 256, 256, 0, stream>>>(
        h, out + (size_t)T_STEPS * BATCH * ODIM, BATCH * HDIM);
}
